// AllGroupsExpertRunner_78288663872352
// MI455X (gfx1250) — compile-verified, matrix operand order checked by reading, not run on hardware
//
#include <hip/hip_runtime.h>
#include <hip/hip_bf16.h>

// MoE SwiGLU FFN for MI455X (gfx1250), wave32 + v_wmma_f32_16x16x32_bf16.
// Pipeline: route(top2 compaction) -> prefix -> cvt tokens bf16 ->
// pack weights into WMMA-fragment layout -> fused gather/SwiGLU/2-GEMM
// (token gather via global_load_async_to_lds_b128, ASYNCcnt-tracked,
//  64-token tile in 72KB LDS to exploit CDNA5's 320KB WGP LDS) ->
// deterministic combine (no float atomics).

typedef __attribute__((ext_vector_type(16))) __bf16 v16bf;
typedef __attribute__((ext_vector_type(8)))  float  v8f;

#define NE   8
#define DD   512
#define HH   2048
#define MM   8192      // B*N
#define TOPK 2
#define MT   64        // token rows per workgroup tile

__device__ __forceinline__ v8f wmma_bf16(v16bf a, v16bf b, v8f c) {
  // (neg_a, A, neg_b, B, c_mod, C, reuse_a, reuse_b)
  return __builtin_amdgcn_wmma_f32_16x16x32_bf16(false, a, false, b, (short)0, c, false, false);
}

// ---------------- routing: compact token lists per expert -----------------
__global__ void k_route(const float* __restrict__ comb, unsigned* __restrict__ cnt,
                        unsigned* __restrict__ list, float* __restrict__ wts,
                        unsigned* __restrict__ pos) {
  int m = blockIdx.x * blockDim.x + threadIdx.x;
  if (m >= MM) return;
  unsigned p0 = 0x80000000u, p1 = 0x80000000u;
  int k = 0;
  for (int e = 0; e < NE; ++e) {
    float w = comb[m * NE + e];
    if (w > 0.0f && k < TOPK) {
      unsigned slot = atomicAdd(&cnt[e], 1u);
      list[(size_t)e * MM + slot] = (unsigned)m;
      wts[(size_t)e * MM + slot]  = w;
      unsigned pe = ((unsigned)e << 28) | slot;
      if (k == 0) p0 = pe; else p1 = pe;
      ++k;
    }
  }
  pos[2 * m]     = p0;
  pos[2 * m + 1] = p1;
}

__global__ void k_prefix(const unsigned* __restrict__ cnt, unsigned* __restrict__ offs) {
  if (threadIdx.x == 0) {
    unsigned s = 0;
    for (int e = 0; e < NE; ++e) { offs[e] = s; s += cnt[e]; }
  }
}

// ---------------- fp32 -> bf16 token conversion ---------------------------
__global__ void k_cvt_x(const float* __restrict__ x, __bf16* __restrict__ xb) {
  int i = blockIdx.x * blockDim.x + threadIdx.x;   // exactly MM*DD threads
  xb[i] = (__bf16)x[i];
}

// ------- pack Wg/Wv/Wo into per-fragment bf16 layout ----------------------
// B-matrix (K x N): block(nt,kt) of 32x16, stored as blk = nt*(K/32)+kt,
// 512 elems per block, per-lane 16 contiguous bf16 (32B) => wave-contiguous 1KB.
__global__ void k_pack_w(const float* __restrict__ Wg, const float* __restrict__ Wv,
                         const float* __restrict__ Wo, __bf16* __restrict__ wgp,
                         __bf16* __restrict__ wvp, __bf16* __restrict__ wop) {
  long long id = (long long)blockIdx.x * blockDim.x + threadIdx.x; // 3*NE*DD*HH
  const long long per = (long long)NE * DD * HH;
  int mat = (int)(id / per);
  long long rem = id - (long long)mat * per;
  int e = (int)(rem / (DD * HH));
  int o = (int)(rem % (DD * HH));
  int j  = o & 15;
  int l  = (o >> 4) & 31;
  int blk = o >> 9;
  int K = (mat == 2) ? HH : DD;
  int N = (mat == 2) ? DD : HH;
  int kblocks = K >> 5;
  int kt = blk % kblocks;
  int nt = blk / kblocks;
  int kk = kt * 32 + ((l >> 4) << 4) + j;   // lanes 16..31 hold K upper half
  int nn = nt * 16 + (l & 15);
  const float* src = (mat == 0) ? Wg : (mat == 1) ? Wv : Wo;
  __bf16* dst      = (mat == 0) ? wgp : (mat == 1) ? wvp : wop;
  dst[(long long)e * (DD * HH) + o] =
      (__bf16)src[(long long)e * (DD * HH) + (long long)kk * N + nn];
}

// ---------------- fused SwiGLU expert FFN (WMMA) --------------------------
// grid = (MM/MT tiles, NE experts), 256 threads = 8 wave32.
// Per H-block(64): gate/value GEMM over K=D=512, gelu*value -> bf16 LDS,
// second GEMM (K=64) accumulates 64x512 output in persistent C-fragments.
__global__ void __launch_bounds__(256) k_moe_ffn(
    const __bf16* __restrict__ xb,  const __bf16* __restrict__ wgp,
    const __bf16* __restrict__ wvp, const __bf16* __restrict__ wop,
    const unsigned* __restrict__ cnt, const unsigned* __restrict__ offs,
    const unsigned* __restrict__ list, const float* __restrict__ wts,
    const float* __restrict__ scales, float* __restrict__ oute) {
  __shared__ __bf16 Xs[MT * DD];     // 64 KB gathered tokens (bf16)
  __shared__ __bf16 gvs[MT * 64];    //  8 KB SwiGLU activations for one H-block
  __shared__ float  wt_s[MT];

  const int e = blockIdx.y;
  const unsigned c  = cnt[e];
  const unsigned m0 = blockIdx.x * MT;
  if (m0 >= c) return;               // uniform early exit

  const int tid = threadIdx.x;

  // Gather MT token rows into LDS: per-lane async DMA (ASYNCcnt-tracked).
  // Each iteration moves one 16B chunk: global x_bf16[tok] row -> LDS tile.
  {
    const unsigned lds_base = (unsigned)(size_t)(&Xs[0]);
    for (int i = tid; i < MT * 64; i += 256) {
      int r = i >> 6, ck = i & 63;
      unsigned mr = m0 + (unsigned)r;
      unsigned lds_addr = lds_base + (unsigned)i * 16u;
      if (mr < c) {
        unsigned tok = list[(size_t)e * MM + mr];
        unsigned long long ga =
            (unsigned long long)(size_t)(xb + (size_t)tok * DD) + (unsigned)ck * 16u;
        asm volatile("global_load_async_to_lds_b128 %0, %1, off"
                     :: "v"(lds_addr), "v"(ga) : "memory");
      } else {
        ((uint4*)Xs)[i] = make_uint4(0u, 0u, 0u, 0u);  // zero-pad tail rows
      }
    }
    for (int r = tid; r < MT; r += 256) {
      unsigned mr = m0 + (unsigned)r;
      wt_s[r] = (mr < c) ? wts[(size_t)e * MM + mr] : 0.0f;
    }
  }
  asm volatile("s_wait_asynccnt 0x0" ::: "memory");
  __syncthreads();

  const int wv = tid >> 5;           // wave id (8 waves)
  const int l  = tid & 31;           // lane
  const int lm = l & 15, lh = l >> 4;

  const int mt1 = wv & 3, ntp = wv >> 2;   // GEMM1: 4 Mtiles x (2 Ntiles each)
  const int mt2 = wv & 3, dh  = wv >> 2;   // GEMM2: 4 Mtiles x 2 D-halves

  const __bf16* wg_e = wgp + (size_t)e * (DD * HH);
  const __bf16* wv_e = wvp + (size_t)e * (DD * HH);
  const __bf16* wo_e = wop + (size_t)e * (DD * HH);

  v8f zero = {};
  v8f co[16];
#pragma unroll
  for (int j = 0; j < 16; ++j) co[j] = zero;  // persistent 64x512 output acc

  for (int hb = 0; hb < 32; ++hb) {
    // ---- GEMM1: gate & value, 2 Ntiles per wave, K = 512 ----
    v8f cg0 = zero, cv0 = zero, cg1 = zero, cv1 = zero;
    const int nta = hb * 4 + ntp * 2;        // H ntile pair (nta, nta+1)
#pragma unroll 4
    for (int kt = 0; kt < 16; ++kt) {
      v16bf a   = *(const v16bf*)(Xs + (mt1 * 16 + lm) * DD + kt * 32 + lh * 16);
      v16bf bg0 = *(const v16bf*)(wg_e + ((size_t)((nta + 0) * 16 + kt)) * 512 + l * 16);
      v16bf bv0 = *(const v16bf*)(wv_e + ((size_t)((nta + 0) * 16 + kt)) * 512 + l * 16);
      v16bf bg1 = *(const v16bf*)(wg_e + ((size_t)((nta + 1) * 16 + kt)) * 512 + l * 16);
      v16bf bv1 = *(const v16bf*)(wv_e + ((size_t)((nta + 1) * 16 + kt)) * 512 + l * 16);
      cg0 = wmma_bf16(a, bg0, cg0);
      cv0 = wmma_bf16(a, bv0, cv0);
      cg1 = wmma_bf16(a, bg1, cg1);
      cv1 = wmma_bf16(a, bv1, cv1);
    }
    __syncthreads();   // previous GEMM2 done reading gvs
    // ---- SwiGLU: gelu(gate)*value, lanes align across cg/cv fragments ----
#pragma unroll
    for (int r = 0; r < 8; ++r) {
      float g0  = cg0[r];
      float gl0 = 0.5f * g0 * (1.0f + erff(g0 * 0.70710678118f));
      gvs[(mt1 * 16 + r + lh * 8) * 64 + (ntp * 2 + 0) * 16 + lm] = (__bf16)(gl0 * cv0[r]);
      float g1  = cg1[r];
      float gl1 = 0.5f * g1 * (1.0f + erff(g1 * 0.70710678118f));
      gvs[(mt1 * 16 + r + lh * 8) * 64 + (ntp * 2 + 1) * 16 + lm] = (__bf16)(gl1 * cv1[r]);
    }
    __syncthreads();
    // ---- GEMM2: out += gv(64x64) @ Wo-block(64x512), K = 64 ----
#pragma unroll
    for (int kt2 = 0; kt2 < 2; ++kt2) {
      v16bf a2 = *(const v16bf*)(gvs + (mt2 * 16 + lm) * 64 + kt2 * 32 + lh * 16);
      const int ktg = hb * 2 + kt2;
#pragma unroll
      for (int j = 0; j < 16; ++j) {
        const int nt = dh * 16 + j;
        v16bf b = *(const v16bf*)(wo_e + ((size_t)(nt * 64 + ktg)) * 512 + l * 16);
        co[j] = wmma_bf16(a2, b, co[j]);
      }
    }
  }

  // Epilogue: fold combine-weight * expert scale, store per-slot rows.
  const float esc = scales[e];
  const unsigned rowbase = offs[e];
#pragma unroll
  for (int j = 0; j < 16; ++j) {
    const int n = (dh * 16 + j) * 16 + lm;
#pragma unroll
    for (int r = 0; r < 8; ++r) {
      const int m = mt2 * 16 + r + lh * 8;
      const unsigned mr = m0 + (unsigned)m;
      if (mr < c) {
        oute[(size_t)(rowbase + mr) * DD + n] = co[j][r] * wt_s[m] * esc;
      }
    }
  }
}

// ---------------- deterministic combine (gather top-2 rows) ---------------
__global__ void k_combine(const float* __restrict__ oute, const unsigned* __restrict__ pos,
                          const unsigned* __restrict__ offs, float* __restrict__ out) {
  int i = blockIdx.x * blockDim.x + threadIdx.x;  // MM*DD/4
  int m  = i >> 7;
  int c4 = i & 127;
  float4 acc = make_float4(0.f, 0.f, 0.f, 0.f);
#pragma unroll
  for (int k = 0; k < TOPK; ++k) {
    unsigned p = pos[2 * m + k];
    if (!(p & 0x80000000u)) {
      unsigned e = p >> 28;
      unsigned slot = p & 0x0FFFFFFFu;
      size_t row = (size_t)offs[e] + slot;
      float4 v = ((const float4*)oute)[row * (DD / 4) + c4];
      acc.x += v.x; acc.y += v.y; acc.z += v.z; acc.w += v.w;
    }
  }
  ((float4*)out)[i] = acc;
}

extern "C" void kernel_launch(void* const* d_in, const int* in_sizes, int n_in,
                              void* d_out, int out_size, void* d_ws, size_t ws_size,
                              hipStream_t stream) {
  (void)in_sizes; (void)n_in; (void)out_size; (void)ws_size;
  const float* tokens = (const float*)d_in[0];
  // d_in[1] = dispatch_weights (mask implied by combine_weights > 0)
  const float* comb   = (const float*)d_in[2];
  const float* Wg     = (const float*)d_in[3];
  const float* Wv     = (const float*)d_in[4];
  const float* Wo     = (const float*)d_in[5];
  const float* scales = (const float*)d_in[6];
  float* out = (float*)d_out;

  char* w = (char*)d_ws;
  size_t off = 0;
  auto take = [&](size_t bytes) -> void* {
    off = (off + 255) & ~(size_t)255;
    void* p = w + off;
    off += bytes;
    return p;
  };
  unsigned* cnt  = (unsigned*)take(NE * 4);
  unsigned* offs = (unsigned*)take(NE * 4);
  unsigned* list = (unsigned*)take((size_t)NE * MM * 4);
  float*    wts  = (float*)   take((size_t)NE * MM * 4);
  unsigned* pos  = (unsigned*)take((size_t)MM * TOPK * 4);
  __bf16*   xb   = (__bf16*)  take((size_t)MM * DD * 2);
  __bf16*   wgp  = (__bf16*)  take((size_t)NE * DD * HH * 2);
  __bf16*   wvp  = (__bf16*)  take((size_t)NE * DD * HH * 2);
  __bf16*   wop  = (__bf16*)  take((size_t)NE * HH * DD * 2);
  float*    oute = (float*)   take((size_t)MM * TOPK * DD * 4);

  hipMemsetAsync(cnt, 0, NE * 4, stream);
  k_route<<<MM / 256, 256, 0, stream>>>(comb, cnt, list, wts, pos);
  k_prefix<<<1, 32, 0, stream>>>(cnt, offs);
  k_cvt_x<<<(MM * DD) / 256, 256, 0, stream>>>(tokens, xb);
  k_pack_w<<<(int)((3LL * NE * DD * HH) / 256), 256, 0, stream>>>(Wg, Wv, Wo, wgp, wvp, wop);
  dim3 g(MM / MT, NE);
  k_moe_ffn<<<g, 256, 0, stream>>>(xb, wgp, wvp, wop, cnt, offs, list, wts, scales, oute);
  k_combine<<<(MM * DD / 4) / 256, 256, 0, stream>>>(oute, pos, offs, out);
}
